// SpatialScaledDotProductforCrossAttention_6803228197062
// MI455X (gfx1250) — matrix operand order repaired and checked by reading, not compiled, not run on hardware
//
#include <hip/hip_runtime.h>
#include <hip/hip_bf16.h>

// Problem constants from the reference
constexpr int Bb  = 4;
constexpr int Hh  = 8;
constexpr int Tt  = 12;
constexpr int Nn  = 325;
constexpr int DK  = 32;
constexpr int BHT = Bb * Hh * Tt;              // 384
constexpr int ROWT   = (Nn + 15) / 16;         // 21 row tiles of 16
constexpr int ROWT2  = (ROWT + 1) / 2;         // 11 row-pair tiles of 32
constexpr int CHUNKS = (Nn + 31) / 32;         // 11 key chunks of 32
constexpr int WPB    = 4;                      // waves per block
constexpr float SCALE = 0.17677669529663687f;  // 1/sqrt(32), folded into Q

typedef __attribute__((ext_vector_type(16))) _Float16 v16h;
typedef __attribute__((ext_vector_type(8)))  _Float16 h8;
typedef __attribute__((ext_vector_type(8)))  float    v8f;
typedef __attribute__((ext_vector_type(2)))  _Float16 h2;

struct LastTag  { static constexpr bool v = true;  };
struct MainTag  { static constexpr bool v = false; };

static __device__ inline h2 shfl_xor_h2(h2 x, int d) {
    int i = __builtin_bit_cast(int, x);
    i = __shfl_xor(i, d, 16);
    return __builtin_bit_cast(h2, i);
}

__global__ __launch_bounds__(WPB * 32) void
spatial_cross_attn_kernel(const float* __restrict__ Qf,
                          const float* __restrict__ Kf,
                          const float* __restrict__ Vf,
                          const float* __restrict__ Qs,
                          const float* __restrict__ Ks,
                          const float* __restrict__ Vs,
                          const float* __restrict__ Kj,
                          const float* __restrict__ Vfp,
                          float* __restrict__ out) {
    // P staging (C-layout -> A-layout hop). Row stride 40 halves = 80B:
    // 16B aligned for ds_load_b128, and row*20 dwords mod 64 yields starts
    // spaced exactly 4 apart -> the b128 read hits all 64 banks once.
    __shared__ __align__(16) _Float16 lds[WPB][2][16][40];

    const int w    = threadIdx.x >> 5;   // wave in block
    const int lane = threadIdx.x & 31;
    const int half = lane >> 4;          // 0: lanes 0-15, 1: lanes 16-31
    const int l16  = lane & 15;

    // One wave = one (variant, bht, 32-row pair of 16-row tiles)
    const int work    = blockIdx.x * WPB + w;
    const int variant = work / (BHT * ROWT2);
    const int rem     = work % (BHT * ROWT2);
    const int bht     = rem / ROWT2;
    const int rowBase = (rem % ROWT2) * 32;

    const float* Q; const float* K; const float* V; bool fs = false;
    if (variant == 0)      { Q = Qf; K = Kf; V = Vf; }
    else if (variant == 1) { Q = Kf; K = Qs; V = Vf; fs = true; }
    else if (variant == 2) { Q = Ks; K = Qf; V = Vs; }
    else                   { Q = Qs; K = Ks; V = Vs; }

    const size_t base = (size_t)bht * Nn * DK;

    // ---- Two Q tiles (16x32 each), A-matrix layout, 1/sqrt(dk) folded ----
    v16h aq[2];
#pragma unroll
    for (int t = 0; t < 2; ++t) {
        const int qrow  = rowBase + t * 16 + l16;
        const int qrowc = qrow < Nn ? qrow : Nn - 1;
        const float* qp = Q + base + (size_t)qrowc * DK;
#pragma unroll
        for (int j = 0; j < 8; ++j) {
            const int kb = 2 * j + 16 * half;
            const float2 tt = *(const float2*)(qp + kb);
            aq[t][2 * j]     = (_Float16)(tt.x * SCALE);
            aq[t][2 * j + 1] = (_Float16)(tt.y * SCALE);
        }
    }

    // All-ones B matrix: rowsum(P) = P @ ones -> softmax denominator in
    // C-layout on every lane (no cross-lane reductions needed for the sum).
    v16h bones;
#pragma unroll
    for (int i = 0; i < 16; ++i) bones[i] = (_Float16)1.0f;

    // K tile (16 keys x 32 dk), B-matrix layout; FlowSpeed transform for v1.
    auto loadB_K = [&](int keybase, auto tag) -> v16h {
        constexpr bool LAST = decltype(tag)::v;
        const int key  = keybase + l16;
        const int keyc = LAST ? (key < Nn ? key : Nn - 1) : key;
        const float* kp = K + base + (size_t)keyc * DK;
        float kj = 1.0f, vfp = 1.0f;
        if (fs) { kj = Kj[keyc]; vfp = Vfp[keyc] + 1e-5f; }
        v16h b;
#pragma unroll
        for (int j = 0; j < 8; ++j) {
            const int kb = 2 * j + 16 * half;
            const float2 tt = *(const float2*)(kp + kb);
            float x = tt.x, y = tt.y;
            if (fs) { x = kj * (x - x * x / vfp); y = kj * (y - y * y / vfp); }
            b[2 * j]     = (_Float16)x;
            b[2 * j + 1] = (_Float16)y;
        }
        return b;
    };

    // V tile (32 keys x 16 dk), B-matrix layout: lane col = dk(dkbase+l16)
    auto loadB_V = [&](int kc0, int dkbase, auto tag) -> v16h {
        constexpr bool LAST = decltype(tag)::v;
        const int dkc = dkbase + l16;
        v16h b;
#pragma unroll
        for (int j = 0; j < 8; ++j) {
            const int kk = 2 * j + 16 * half;
            int k0 = kc0 + kk;
            int k1 = kc0 + kk + 1;
            if (LAST) { k0 = k0 < Nn ? k0 : Nn - 1; k1 = k1 < Nn ? k1 : Nn - 1; }
            b[2 * j]     = (_Float16)V[base + (size_t)k0 * DK + dkc];
            b[2 * j + 1] = (_Float16)V[base + (size_t)k1 * DK + dkc];
        }
        return b;
    };

    // Online-softmax state per tile. C-layout rows: row = r + 8*half.
    float m[2][8], ls[2][8];
    v8f o0[2] = {{}, {}}, o1[2] = {{}, {}};
#pragma unroll
    for (int t = 0; t < 2; ++t)
#pragma unroll
        for (int r = 0; r < 8; ++r) { m[t][r] = -1e30f; ls[t][r] = 0.0f; }

    auto chunk_body = [&](int kc0, auto tag) {
        constexpr bool LAST = decltype(tag)::v;

        // S = Q @ K^T: shared B operands (bk0,bk1) across both A tiles
        const v16h bk0 = loadB_K(kc0, tag);
        const v16h bk1 = loadB_K(kc0 + 16, tag);
        v8f z = {};
        v8f s0[2], s1[2];
#pragma unroll
        for (int t = 0; t < 2; ++t) {
            s0[t] = __builtin_amdgcn_wmma_f32_16x16x32_f16(false, aq[t], false, bk0,
                                                           (short)0, z, false, false);
            s1[t] = __builtin_amdgcn_wmma_f32_16x16x32_f16(false, aq[t], false, bk1,
                                                           (short)0, z, false, false);
        }

        float corr[2][8];
#pragma unroll
        for (int t = 0; t < 2; ++t) {
            float t0[8], t1[8];
#pragma unroll
            for (int r = 0; r < 8; ++r) {
                float x0 = s0[t][r], x1 = s1[t][r];
                if (LAST) {       // only the peeled tail chunk pays for masking
                    x0 = (kc0 + l16 < Nn)      ? x0 : -1e30f;
                    x1 = (kc0 + 16 + l16 < Nn) ? x1 : -1e30f;
                }
                t0[r] = x0; t1[r] = x1;
            }
            // Row-max: packed-f16 half-wave butterfly. Softmax is shift-
            // invariant, so the f16-rounded max (<=~0.01 shift) is safe.
            h2 hm[4];
#pragma unroll
            for (int i = 0; i < 4; ++i) {
                h2 p;
                p.x = (_Float16)fmaxf(t0[2 * i],     t1[2 * i]);
                p.y = (_Float16)fmaxf(t0[2 * i + 1], t1[2 * i + 1]);
                hm[i] = p;
            }
#pragma unroll
            for (int d = 1; d < 16; d <<= 1) {
#pragma unroll
                for (int i = 0; i < 4; ++i)
                    hm[i] = __builtin_elementwise_max(hm[i], shfl_xor_h2(hm[i], d));
            }
#pragma unroll
            for (int r = 0; r < 8; ++r) {
                const float mx = (float)((r & 1) ? hm[r >> 1].y : hm[r >> 1].x);
                const float mnew = fmaxf(m[t][r], mx);
                corr[t][r] = __expf(m[t][r] - mnew);
                m[t][r] = mnew;
                const float e0 = __expf(t0[r] - mnew);  // ->0 if masked
                const float e1 = __expf(t1[r] - mnew);
                o0[t][r] *= corr[t][r];
                o1[t][r] *= corr[t][r];
                const int row = r + 8 * half;           // C-layout row
                lds[w][t][row][l16]      = (_Float16)e0;
                lds[w][t][row][16 + l16] = (_Float16)e1;
            }
        }
        __syncthreads();

        // Shared B operands for the PV GEMM
        const v16h bv0 = loadB_V(kc0, 0, tag);
        const v16h bv1 = loadB_V(kc0, 16, tag);

#pragma unroll
        for (int t = 0; t < 2; ++t) {
            // A-layout P: two contiguous 16B runs -> 2x ds_load_b128
            const h8 lo = *(const h8*)&lds[w][t][l16][16 * half];
            const h8 hi = *(const h8*)&lds[w][t][l16][16 * half + 8];
            const v16h pa = __builtin_shufflevector(lo, hi,
                0, 1, 2, 3, 4, 5, 6, 7, 8, 9, 10, 11, 12, 13, 14, 15);

            // Denominator via WMMA: rowsum(P) broadcast across C-layout lanes
            v8f ssum = __builtin_amdgcn_wmma_f32_16x16x32_f16(
                false, pa, false, bones, (short)0, z, false, false);
            o0[t] = __builtin_amdgcn_wmma_f32_16x16x32_f16(
                false, pa, false, bv0, (short)0, o0[t], false, false);
            o1[t] = __builtin_amdgcn_wmma_f32_16x16x32_f16(
                false, pa, false, bv1, (short)0, o1[t], false, false);
#pragma unroll
            for (int r = 0; r < 8; ++r)
                ls[t][r] = ls[t][r] * corr[t][r] + ssum[r];
        }
        __syncthreads();
    };

    // Main chunks 0..9 cover keys 0..319 < 325: no clamps, no masking.
    for (int c = 0; c < CHUNKS - 1; ++c)
        chunk_body(c * 32, MainTag{});
    chunk_body((CHUNKS - 1) * 32, LastTag{});   // peeled masked tail

    // Normalize and store (C-layout: row = r + 8*half, dk col = l16 / l16+16)
    const size_t outbase = (size_t)variant * BHT * Nn * DK + base;
#pragma unroll
    for (int t = 0; t < 2; ++t) {
#pragma unroll
        for (int r = 0; r < 8; ++r) {
            const int row = rowBase + t * 16 + r + 8 * half;
            if (row < Nn) {
                const float inv = 1.0f / ls[t][r];
                out[outbase + (size_t)row * DK + l16]      = o0[t][r] * inv;
                out[outbase + (size_t)row * DK + 16 + l16] = o1[t][r] * inv;
            }
        }
    }
}

extern "C" void kernel_launch(void* const* d_in, const int* in_sizes, int n_in,
                              void* d_out, int out_size, void* d_ws, size_t ws_size,
                              hipStream_t stream) {
    (void)in_sizes; (void)n_in; (void)out_size; (void)d_ws; (void)ws_size;
    const float* Qf  = (const float*)d_in[0];
    const float* Kf  = (const float*)d_in[1];
    const float* Vf  = (const float*)d_in[2];
    const float* Qs  = (const float*)d_in[3];
    const float* Ks  = (const float*)d_in[4];
    const float* Vs  = (const float*)d_in[5];
    const float* Kj  = (const float*)d_in[6];
    const float* Vfp = (const float*)d_in[7];
    float* out = (float*)d_out;

    const int totalWaves = 4 * BHT * ROWT2;         // 16896
    const int blocks     = totalWaves / WPB;        // 4224
    spatial_cross_attn_kernel<<<blocks, WPB * 32, 0, stream>>>(
        Qf, Kf, Vf, Qs, Ks, Vs, Kj, Vfp, out);
}
